// Attention_55825984913972
// MI455X (gfx1250) — matrix operand order repaired and checked by reading, not compiled, not run on hardware
//
#include <hip/hip_runtime.h>
#include <hip/hip_bf16.h>

typedef _Float16 half8  __attribute__((ext_vector_type(8)));
typedef _Float16 half16 __attribute__((ext_vector_type(16)));
typedef float    v8f    __attribute__((ext_vector_type(8)));

#define SEQ   4096
#define DIM   1024
#define NQKV  3072
#define HEADS 16
#define DHEAD 64

// ---------------------------------------------------------------------------
// Fragment helpers (wave32 WMMA layouts per CDNA5 ISA 7.12.2)
//  A 16x32 f16: lane L: M=L&15; halves 0..7 -> K = 8*hs+0..7 ; halves 8..15 -> K=16+8*hs+0..7
//  B 32x16 f16: lane L: N=L&15; halves 0..7 -> K = 8*hs+0..7 ; halves 8..15 -> K=16+8*hs+0..7
//  C/D 16x16 f32: VGPR v: row = v + 8*hs, col = L&15
// ---------------------------------------------------------------------------
__device__ __forceinline__ half16 frag_a(const _Float16* base, int stride, int lane) {
    int r = lane & 15, hs = lane >> 4;
    const half8* p = (const half8*)(base + r * stride + hs * 8);
    union { half16 f; half8 h[2]; } u;
    u.h[0] = p[0];   // K chunk at +8*hs
    u.h[1] = p[2];   // K chunk at +16+8*hs
    return u.f;
}

__device__ __forceinline__ half16 frag_b(const _Float16* base, int stride, int lane) {
    int r = lane & 15, hs = lane >> 4;
    const half8* p = (const half8*)(base + r * stride + hs * 8);
    union { half16 f; half8 h[2]; } u;
    u.h[0] = p[0];
    u.h[1] = p[2];   // K chunk at +16+8*hs
    return u.f;
}

__device__ __forceinline__ v8f wmma_f16(half16 a, half16 b, v8f c) {
    return __builtin_amdgcn_wmma_f32_16x16x32_f16(false, a, false, b, (short)0, c, false, false);
}

// CDNA5 async copy: global -> LDS, 16B per lane, tracked by ASYNCcnt.
// GVS addressing: saddr (SGPR pair, uniform base) + 32-bit per-lane byte offset.
// lds_addr = low 32 bits of flat shared address (== wave-relative LDS offset
// per the aperture mapping).
__device__ __forceinline__ void async_b128(unsigned lds_addr, int gvoff_bytes, const void* gbase) {
    asm volatile("global_load_async_to_lds_b128 %0, %1, %2"
                 :: "v"(lds_addr), "v"(gvoff_bytes), "s"(gbase)
                 : "memory");
}
#define ASYNC_WAIT(n) asm volatile("s_wait_asynccnt %0" :: "i"(n) : "memory")

// ---------------------------------------------------------------------------
// Kernel 1: weight convert fp32->f16 with transpose (B tiles become K-major)
// ---------------------------------------------------------------------------
__global__ __launch_bounds__(256) void weight_prep_kernel(
    const float* __restrict__ w_qkv, const float* __restrict__ w_out,
    _Float16* __restrict__ wqkvT, _Float16* __restrict__ woutT) {
    int idx = blockIdx.x * 256 + threadIdx.x;
    const int T1 = NQKV * DIM;           // 3072*1024
    if (idx < T1) {
        int n = idx >> 10, k = idx & 1023;
        wqkvT[idx] = (_Float16)w_qkv[k * NQKV + n];
    } else if (idx < T1 + DIM * DIM) {
        int j = idx - T1;
        int n = j >> 10, k = j & 1023;
        woutT[j] = (_Float16)w_out[k * DIM + n];
    }
}

// ---------------------------------------------------------------------------
// Kernel 2: LayerNorm (fp32 in) -> f16 normalized activations
// ---------------------------------------------------------------------------
__global__ __launch_bounds__(256) void ln_kernel(
    const float* __restrict__ x, const float* __restrict__ g,
    const float* __restrict__ b, _Float16* __restrict__ y) {
    int row = blockIdx.x, tid = threadIdx.x;
    const float4 xv = ((const float4*)(x + row * DIM))[tid];
    float s  = xv.x + xv.y + xv.z + xv.w;
    float s2 = xv.x * xv.x + xv.y * xv.y + xv.z * xv.z + xv.w * xv.w;
    __shared__ float rs[256], rs2[256];
    rs[tid] = s; rs2[tid] = s2;
    __syncthreads();
    for (int off = 128; off > 0; off >>= 1) {
        if (tid < off) { rs[tid] += rs[tid + off]; rs2[tid] += rs2[tid + off]; }
        __syncthreads();
    }
    float mean = rs[0] * (1.0f / DIM);
    float var  = rs2[0] * (1.0f / DIM) - mean * mean;
    float rstd = rsqrtf(var + 1e-6f);
    float4 gv = ((const float4*)g)[tid];
    float4 bv = ((const float4*)b)[tid];
    int o = row * DIM + tid * 4;
    y[o + 0] = (_Float16)((xv.x - mean) * rstd * gv.x + bv.x);
    y[o + 1] = (_Float16)((xv.y - mean) * rstd * gv.y + bv.y);
    y[o + 2] = (_Float16)((xv.z - mean) * rstd * gv.z + bv.z);
    y[o + 3] = (_Float16)((xv.w - mean) * rstd * gv.w + bv.w);
}

// ---------------------------------------------------------------------------
// Kernel 3: QKV GEMM  [4096,1024] x [1024,3072] -> q_h, k_h, vT (all f16)
// 256 thr (8 waves), 128x128 tile, K-chunks of 32, LDS **double-buffered via
// global_load_async_to_lds_b128** (ASYNCcnt-pipelined). Wave grid 4(M)x2(N).
// V is stored transposed: vT[(col-2048)*SEQ + row].
// ---------------------------------------------------------------------------
__global__ __launch_bounds__(256) void qkv_gemm_kernel(
    const _Float16* __restrict__ A, const _Float16* __restrict__ Bt,
    _Float16* __restrict__ qh, _Float16* __restrict__ kh, _Float16* __restrict__ vT) {
    __shared__ __align__(16) _Float16 Asm[2][128][40];
    __shared__ __align__(16) _Float16 Bsm[2][128][40];
    int tid = threadIdx.x, lane = tid & 31, wid = tid >> 5;
    int m0 = blockIdx.x * 128, n0 = blockIdx.y * 128;
    int wm = wid & 3, wn = wid >> 2;
    int r = lane & 15, hs = lane >> 4;

    auto issue = [&](int buf, int k0) {
#pragma unroll
        for (int i = 0; i < 2; ++i) {
            int c = tid + i * 256;            // 0..511
            int row = c >> 2, ch = (c & 3) * 8;
            unsigned la = (unsigned)(size_t)&Asm[buf][row][ch];
            unsigned lb = (unsigned)(size_t)&Bsm[buf][row][ch];
            async_b128(la, ((m0 + row) * DIM + k0 + ch) * 2, A);
            async_b128(lb, ((n0 + row) * DIM + k0 + ch) * 2, Bt);
        }
    };

    v8f acc[2][4];
#pragma unroll
    for (int s = 0; s < 2; ++s)
#pragma unroll
        for (int t = 0; t < 4; ++t) acc[s][t] = (v8f)0.0f;

    issue(0, 0);
    for (int k0 = 0; k0 < DIM; k0 += 32) {
        int buf = (k0 >> 5) & 1;
        if (k0 + 32 < DIM) { issue(buf ^ 1, k0 + 32); ASYNC_WAIT(4); }
        else               { ASYNC_WAIT(0); }
        __syncthreads();
        half16 af[2], bf[4];
#pragma unroll
        for (int s = 0; s < 2; ++s) af[s] = frag_a(&Asm[buf][wm * 32 + s * 16][0], 40, lane);
#pragma unroll
        for (int t = 0; t < 4; ++t) bf[t] = frag_b(&Bsm[buf][wn * 64 + t * 16][0], 40, lane);
#pragma unroll
        for (int s = 0; s < 2; ++s)
#pragma unroll
            for (int t = 0; t < 4; ++t) acc[s][t] = wmma_f16(af[s], bf[t], acc[s][t]);
        __syncthreads();
    }

#pragma unroll
    for (int s = 0; s < 2; ++s) {
#pragma unroll
        for (int t = 0; t < 4; ++t) {
            int nc0 = n0 + wn * 64 + t * 16;      // tile stays inside one of q/k/v
            int col = nc0 + r;
            int rb  = m0 + wm * 32 + s * 16 + 8 * hs;
            if (nc0 < DIM) {
#pragma unroll
                for (int v = 0; v < 8; ++v) qh[(rb + v) * DIM + col] = (_Float16)acc[s][t][v];
            } else if (nc0 < 2 * DIM) {
#pragma unroll
                for (int v = 0; v < 8; ++v) kh[(rb + v) * DIM + (col - DIM)] = (_Float16)acc[s][t][v];
            } else {
#pragma unroll
                for (int v = 0; v < 8; ++v) vT[(col - 2 * DIM) * SEQ + rb + v] = (_Float16)acc[s][t][v];
            }
        }
    }
}

// ---------------------------------------------------------------------------
// Kernel 4: flash attention. One wave per (head, 16-query tile).
// Per 32-key chunk: S = Q*K^T (4 WMMA), online softmax (shfl over 16-lane row
// groups), P relaid C->A layout through per-wave LDS (in-order DS, no block
// barrier needed), O += P*V (4 WMMA). Next chunks prefetched (global_prefetch).
// ---------------------------------------------------------------------------
__global__ __launch_bounds__(128) void flash_kernel(
    const _Float16* __restrict__ qh, const _Float16* __restrict__ kh,
    const _Float16* __restrict__ vT, _Float16* __restrict__ oh) {
    __shared__ __align__(16) _Float16 pbuf[4][16][40];
    int lane = threadIdx.x & 31, wid = threadIdx.x >> 5;
    int gw = blockIdx.x * 4 + wid;
    int h  = gw >> 8;            // 256 query tiles per head
    int q0 = (gw & 255) * 16;
    int r = lane & 15, hs = lane >> 4;

    half16 aq[2];
#pragma unroll
    for (int d = 0; d < 2; ++d)
        aq[d] = frag_a(qh + (size_t)(q0) * DIM + h * DHEAD + d * 32, DIM, lane);

    float m[8], l[8];
    v8f accO[4];
#pragma unroll
    for (int v = 0; v < 8; ++v) { m[v] = -1e30f; l[v] = 0.0f; }
#pragma unroll
    for (int t = 0; t < 4; ++t) accO[t] = (v8f)0.0f;

    const float sl = 0.03125f * 1.44269504089f;   // dim_in^-0.5 * log2(e)

    for (int j0 = 0; j0 < SEQ; j0 += 32) {
        if (j0 + 64 < SEQ) {   // prefetch next-next K rows and V columns
            __builtin_prefetch(kh + (size_t)(j0 + 64 + lane) * DIM + h * DHEAD, 0, 0);
            __builtin_prefetch(vT + (size_t)(h * DHEAD + lane) * SEQ + j0 + 64, 0, 0);
            __builtin_prefetch(vT + (size_t)(h * DHEAD + 32 + lane) * SEQ + j0 + 64, 0, 0);
        }
        // ---- scores: two 16x16 tiles over keys j0..j0+15 and j0+16..j0+31
        v8f s0 = (v8f)0.0f, s1 = (v8f)0.0f;
#pragma unroll
        for (int d = 0; d < 2; ++d) {
            half16 bk0 = frag_b(kh + (size_t)(j0)      * DIM + h * DHEAD + d * 32, DIM, lane);
            half16 bk1 = frag_b(kh + (size_t)(j0 + 16) * DIM + h * DHEAD + d * 32, DIM, lane);
            s0 = wmma_f16(aq[d], bk0, s0);
            s1 = wmma_f16(aq[d], bk1, s1);
        }
        // ---- online softmax per row (row = v + 8*hs; 16 cols across 16-lane group)
#pragma unroll
        for (int v = 0; v < 8; ++v) {
            float a0 = s0[v] * sl, a1 = s1[v] * sl;
            float rm = fmaxf(a0, a1);
#pragma unroll
            for (int off = 1; off < 16; off <<= 1) rm = fmaxf(rm, __shfl_xor(rm, off, 32));
            float mn    = fmaxf(m[v], rm);
            float alpha = exp2f(m[v] - mn);
            float p0 = exp2f(a0 - mn), p1 = exp2f(a1 - mn);
            float rs = p0 + p1;
#pragma unroll
            for (int off = 1; off < 16; off <<= 1) rs += __shfl_xor(rs, off, 32);
            l[v] = l[v] * alpha + rs;
            m[v] = mn;
            accO[0][v] *= alpha; accO[1][v] *= alpha;
            accO[2][v] *= alpha; accO[3][v] *= alpha;
            int row = v + 8 * hs;
            pbuf[wid][row][r]      = (_Float16)p0;
            pbuf[wid][row][16 + r] = (_Float16)p1;
        }
        // pbuf is wave-private; DS ops complete in order per wave -> no barrier.
        half16 pf = frag_a(&pbuf[wid][0][0], 40, lane);
        // ---- O += P * V  (V transposed: contiguous along j)
#pragma unroll
        for (int t = 0; t < 4; ++t) {
            half16 bv = frag_b(vT + (size_t)(h * DHEAD + t * 16) * SEQ + j0, SEQ, lane);
            accO[t] = wmma_f16(pf, bv, accO[t]);
        }
    }

#pragma unroll
    for (int v = 0; v < 8; ++v) {
        float inv = 1.0f / l[v];
        int row = q0 + v + 8 * hs;
#pragma unroll
        for (int t = 0; t < 4; ++t)
            oh[(size_t)row * DIM + h * DHEAD + t * 16 + r] = (_Float16)(accO[t][v] * inv);
    }
}

// ---------------------------------------------------------------------------
// Kernel 5: output GEMM [4096,1024] x [1024,1024] + bias -> fp32
// Same async double-buffered pipeline as kernel 3.
// ---------------------------------------------------------------------------
__global__ __launch_bounds__(256) void out_gemm_kernel(
    const _Float16* __restrict__ A, const _Float16* __restrict__ Bt,
    const float* __restrict__ bias, float* __restrict__ out) {
    __shared__ __align__(16) _Float16 Asm[2][128][40];
    __shared__ __align__(16) _Float16 Bsm[2][128][40];
    int tid = threadIdx.x, lane = tid & 31, wid = tid >> 5;
    int m0 = blockIdx.x * 128, n0 = blockIdx.y * 128;
    int wm = wid & 3, wn = wid >> 2;
    int r = lane & 15, hs = lane >> 4;

    auto issue = [&](int buf, int k0) {
#pragma unroll
        for (int i = 0; i < 2; ++i) {
            int c = tid + i * 256;
            int row = c >> 2, ch = (c & 3) * 8;
            unsigned la = (unsigned)(size_t)&Asm[buf][row][ch];
            unsigned lb = (unsigned)(size_t)&Bsm[buf][row][ch];
            async_b128(la, ((m0 + row) * DIM + k0 + ch) * 2, A);
            async_b128(lb, ((n0 + row) * DIM + k0 + ch) * 2, Bt);
        }
    };

    v8f acc[2][4];
#pragma unroll
    for (int s = 0; s < 2; ++s)
#pragma unroll
        for (int t = 0; t < 4; ++t) acc[s][t] = (v8f)0.0f;

    issue(0, 0);
    for (int k0 = 0; k0 < DIM; k0 += 32) {
        int buf = (k0 >> 5) & 1;
        if (k0 + 32 < DIM) { issue(buf ^ 1, k0 + 32); ASYNC_WAIT(4); }
        else               { ASYNC_WAIT(0); }
        __syncthreads();
        half16 af[2], bf[4];
#pragma unroll
        for (int s = 0; s < 2; ++s) af[s] = frag_a(&Asm[buf][wm * 32 + s * 16][0], 40, lane);
#pragma unroll
        for (int t = 0; t < 4; ++t) bf[t] = frag_b(&Bsm[buf][wn * 64 + t * 16][0], 40, lane);
#pragma unroll
        for (int s = 0; s < 2; ++s)
#pragma unroll
            for (int t = 0; t < 4; ++t) acc[s][t] = wmma_f16(af[s], bf[t], acc[s][t]);
        __syncthreads();
    }

#pragma unroll
    for (int s = 0; s < 2; ++s) {
#pragma unroll
        for (int t = 0; t < 4; ++t) {
            int col = n0 + wn * 64 + t * 16 + r;
            int rb  = m0 + wm * 32 + s * 16 + 8 * hs;
            float bb = bias[col];
#pragma unroll
            for (int v = 0; v < 8; ++v) out[(rb + v) * DIM + col] = acc[s][t][v] + bb;
        }
    }
}

// ---------------------------------------------------------------------------
extern "C" void kernel_launch(void* const* d_in, const int* in_sizes, int n_in,
                              void* d_out, int out_size, void* d_ws, size_t ws_size,
                              hipStream_t stream) {
    const float* x        = (const float*)d_in[0];
    const float* ln_scale = (const float*)d_in[1];
    const float* ln_bias  = (const float*)d_in[2];
    const float* w_qkv    = (const float*)d_in[3];
    const float* w_out    = (const float*)d_in[4];
    const float* b_out    = (const float*)d_in[5];
    float* out = (float*)d_out;

    _Float16* ws    = (_Float16*)d_ws;
    _Float16* xn    = ws;                         // [4096][1024]
    _Float16* wqkvT = xn    + (size_t)SEQ * DIM;  // [3072][1024]
    _Float16* woutT = wqkvT + (size_t)NQKV * DIM; // [1024][1024]
    _Float16* qh    = woutT + (size_t)DIM * DIM;  // [4096][1024]
    _Float16* kh    = qh    + (size_t)SEQ * DIM;  // [4096][1024]
    _Float16* vT    = kh    + (size_t)SEQ * DIM;  // [1024][4096]  (h*64+d major)
    _Float16* oh    = vT    + (size_t)DIM * SEQ;  // [4096][1024]

    weight_prep_kernel<<<(NQKV * DIM + DIM * DIM) / 256, 256, 0, stream>>>(w_qkv, w_out, wqkvT, woutT);
    ln_kernel<<<SEQ, 256, 0, stream>>>(x, ln_scale, ln_bias, xn);
    qkv_gemm_kernel<<<dim3(SEQ / 128, NQKV / 128), 256, 0, stream>>>(xn, wqkvT, qh, kh, vT);
    flash_kernel<<<(HEADS * (SEQ / 16)) / 4, 128, 0, stream>>>(qh, kh, vT, oh);
    out_gemm_kernel<<<dim3(SEQ / 128, DIM / 128), 256, 0, stream>>>(oh, woutT, b_out, out);
}
